// PrgSelfAtt_26723286516092
// MI455X (gfx1250) — matrix operand order repaired and checked by reading, not compiled, not run on hardware
//
#include <hip/hip_runtime.h>

// ---------------------------------------------------------------------------
// Non-local self-attention block for MI455X (gfx1250), bf16 WMMA pipeline.
// Block tile 64(M)x128(N), BK=32, 128 threads (4 waves, each 32x64 ->
// 8 wmma/K-step), double-buffered LDS in a fragment-swizzled layout
// (every WMMA operand read = 2x ds_load_b128). k_pv and k_outproj-B stage
// tiles with GLOBAL_LOAD_ASYNC_TO_LDS_B128 (ASYNCcnt-tracked async engine).
// ---------------------------------------------------------------------------

#define L_TOT   6272
#define N_B     2
#define C_DIM   1024
#define C_INNER 512
#define EPS_GN  1e-5f
#define SCALE_QK 0.04419417382415922f   // 512^-0.5

typedef __bf16 bf16;
typedef bf16  bf16x4  __attribute__((ext_vector_type(4)));
typedef bf16  bf16x8  __attribute__((ext_vector_type(8)));
typedef bf16  bf16x16 __attribute__((ext_vector_type(16)));
typedef float v8f     __attribute__((ext_vector_type(8)));

#define ASZ 2048   // 64 x 32 bf16 (A tile), one buffer
#define BSZ 4096   // 32 x 128 bf16 (B tile), one buffer

// ---- fragment-swizzled LDS indexing ---------------------------------------
// idx = group*512 + lane*16 + elem; each lane's 16 elems are the exact v16bf
// WMMA operand (K-sequential).
__device__ __forceinline__ int a_sw(int m, int k) {
  return ((m >> 4) << 9) + ((k & 16) + (m & 15)) * 16 + (k & 15);
}
__device__ __forceinline__ int b_sw(int k, int n) {
  return ((n >> 4) << 9) + ((k & 16) + (n & 15)) * 16 + (k & 15);
}
__device__ __forceinline__ bf16x16 frag_ld(const bf16* tile, int group, int lane) {
  return *(const bf16x16*)(tile + (group << 9) + lane * 16);   // 2x ds_load_b128
}

// Async 16B copy global -> LDS (per-lane LDS dest + per-lane global addr).
// Generic LDS pointers keep the wave-relative LDS offset in the low 32 bits.
__device__ __forceinline__ void async_b128(const bf16* lds_dst, const bf16* gsrc) {
  asm volatile("global_load_async_to_lds_b128 %0, %1, off"
               :: "v"((unsigned)(unsigned long long)(uintptr_t)lds_dst),
                  "v"((unsigned long long)(uintptr_t)gsrc)
               : "memory");
}
__device__ __forceinline__ void wait_async0() {
  asm volatile("s_wait_asynccnt 0x0" ::: "memory");
}

// wave computes 32x64: groups wmg..wmg+1 (A) x wng..wng+3 (B), 8 wmma.
__device__ __forceinline__ void mma8(const bf16* As, const bf16* Bs,
                                     v8f acc[2][4], int wmg, int wng, int lane) {
  bf16x16 a0 = frag_ld(As, wmg + 0, lane);
  bf16x16 a1 = frag_ld(As, wmg + 1, lane);
#pragma unroll
  for (int j = 0; j < 4; ++j) {
    bf16x16 b = frag_ld(Bs, wng + j, lane);
    acc[0][j] = __builtin_amdgcn_wmma_f32_16x16x32_bf16(false, a0, false, b, (short)0, acc[0][j], false, false);
    acc[1][j] = __builtin_amdgcn_wmma_f32_16x16x32_bf16(false, a1, false, b, (short)0, acc[1][j], false, false);
  }
}

// ---------------------------------------------------------------------------
// K1: theta/phi/g projections. C[d,l] = sum_c W[d,c] * x[n,c,l] (+bias)
// theta pre-scaled by 512^-0.5. qkv layout: [proj][n][512][L] bf16.
// (f32 sources need conversion -> VGPR staging.)
// ---------------------------------------------------------------------------
__global__ __launch_bounds__(128)
void k_proj3(const float* __restrict__ x,
             const float* __restrict__ wt, const float* __restrict__ bt,
             const float* __restrict__ wp, const float* __restrict__ bp,
             const float* __restrict__ wg, const float* __restrict__ bg,
             bf16* __restrict__ qkv) {
  __shared__ bf16 As[2 * ASZ];
  __shared__ bf16 Bs[2 * BSZ];
  const int l0 = blockIdx.x * 128;
  const int m0 = blockIdx.y * 64;
  const int n  = blockIdx.z / 3, pr = blockIdx.z % 3;
  const float* W; const float* Bias; float scale;
  if (pr == 0)      { W = wt; Bias = bt; scale = SCALE_QK; }
  else if (pr == 1) { W = wp; Bias = bp; scale = 1.f; }
  else              { W = wg; Bias = bg; scale = 1.f; }
  const float* xb = x + (size_t)n * C_DIM * L_TOT;
  bf16* outb = qkv + ((size_t)pr * N_B + n) * (size_t)C_INNER * L_TOT;

  const int tid = threadIdx.x, lane = tid & 31, wave = tid >> 5;
  const int wmg = (wave >> 1) * 2, wng = (wave & 1) * 4;
  const int wm = wmg * 16, wn = wng * 16;
  v8f acc[2][4] = {};

  const int rA = tid >> 3, cA = (tid & 7) * 4;
  const int rB = tid >> 5, cB = (tid & 31) * 4;
  float4 va[4], vb[8];

  auto loadT = [&](int k) {
#pragma unroll
    for (int i = 0; i < 4; ++i)
      va[i] = *(const float4*)(W + (size_t)(m0 + i * 16 + rA) * C_DIM + k + cA);
#pragma unroll
    for (int i = 0; i < 8; ++i)
      vb[i] = *(const float4*)(xb + (size_t)(k + i * 4 + rB) * L_TOT + l0 + cB);
  };
  auto storeT = [&](bf16* as, bf16* bs) {
#pragma unroll
    for (int i = 0; i < 4; ++i) {
      bf16x4 t = { (bf16)va[i].x, (bf16)va[i].y, (bf16)va[i].z, (bf16)va[i].w };
      *(bf16x4*)(as + a_sw(i * 16 + rA, cA)) = t;
    }
#pragma unroll
    for (int i = 0; i < 8; ++i) {
      const int kk = i * 4 + rB;
      bs[b_sw(kk, cB + 0)] = (bf16)vb[i].x;
      bs[b_sw(kk, cB + 1)] = (bf16)vb[i].y;
      bs[b_sw(kk, cB + 2)] = (bf16)vb[i].z;
      bs[b_sw(kk, cB + 3)] = (bf16)vb[i].w;
    }
  };

  const int NK = C_DIM / 32;
  loadT(0); storeT(As, Bs);
  int cur = 0;
  for (int kt = 0; kt < NK; ++kt) {
    __syncthreads();
    if (kt + 1 < NK) loadT((kt + 1) * 32);
    mma8(As + cur * ASZ, Bs + cur * BSZ, acc, wmg, wng, lane);
    if (kt + 1 < NK) { cur ^= 1; storeT(As + cur * ASZ, Bs + cur * BSZ); }
  }

  const int nn = lane & 15, mofs = (lane >> 4) * 8;
#pragma unroll
  for (int i = 0; i < 2; ++i)
#pragma unroll
    for (int j = 0; j < 4; ++j)
#pragma unroll
      for (int r = 0; r < 8; ++r) {
        const int d = m0 + wm + i * 16 + mofs + r;
        const int l = l0 + wn + j * 16 + nn;
        outb[(size_t)d * L_TOT + l] = (bf16)((acc[i][j][r] + Bias[d]) * scale);
      }
}

// ---------------------------------------------------------------------------
// K2: scores S[t,p] = sum_c theta[c,t] * phi[c,p]   (theta pre-scaled)
// (2-byte transposed scatter -> VGPR staging.)
// ---------------------------------------------------------------------------
__global__ __launch_bounds__(128)
void k_scores(const bf16* __restrict__ theta, const bf16* __restrict__ phi,
              float* __restrict__ S) {
  __shared__ bf16 As[2 * ASZ];
  __shared__ bf16 Bs[2 * BSZ];
  const int p0 = blockIdx.x * 128;
  const int t0 = blockIdx.y * 64;
  const int tid = threadIdx.x, lane = tid & 31, wave = tid >> 5;
  const int wmg = (wave >> 1) * 2, wng = (wave & 1) * 4;
  const int wm = wmg * 16, wn = wng * 16;
  v8f acc[2][4] = {};

  const int cAr = tid >> 3, tA = (tid & 7) * 8;
  const int cBr = tid >> 4, pB = (tid & 15) * 8;
  bf16x8 va[2], vb[4];

  auto loadT = [&](int k) {
#pragma unroll
    for (int i = 0; i < 2; ++i)
      va[i] = *(const bf16x8*)(theta + (size_t)(k + i * 16 + cAr) * L_TOT + t0 + tA);
#pragma unroll
    for (int i = 0; i < 4; ++i)
      vb[i] = *(const bf16x8*)(phi + (size_t)(k + i * 8 + cBr) * L_TOT + p0 + pB);
  };
  auto storeT = [&](bf16* as, bf16* bs) {
#pragma unroll
    for (int i = 0; i < 2; ++i) {
      const int cc = i * 16 + cAr;
#pragma unroll
      for (int q = 0; q < 8; ++q) as[a_sw(tA + q, cc)] = va[i][q];   // transpose
    }
#pragma unroll
    for (int i = 0; i < 4; ++i) {
      const int cc = i * 8 + cBr;
#pragma unroll
      for (int q = 0; q < 8; ++q) bs[b_sw(cc, pB + q)] = vb[i][q];
    }
  };

  const int NK = C_INNER / 32;
  loadT(0); storeT(As, Bs);
  int cur = 0;
  for (int kt = 0; kt < NK; ++kt) {
    __syncthreads();
    if (kt + 1 < NK) loadT((kt + 1) * 32);
    mma8(As + cur * ASZ, Bs + cur * BSZ, acc, wmg, wng, lane);
    if (kt + 1 < NK) { cur ^= 1; storeT(As + cur * ASZ, Bs + cur * BSZ); }
  }

  const int nn = lane & 15, mofs = (lane >> 4) * 8;
#pragma unroll
  for (int i = 0; i < 2; ++i)
#pragma unroll
    for (int j = 0; j < 4; ++j)
#pragma unroll
      for (int r = 0; r < 8; ++r) {
        const int t = t0 + wm + i * 16 + mofs + r;
        const int p = p0 + wn + j * 16 + nn;
        S[(size_t)t * L_TOT + p] = acc[i][j][r];
      }
}

// ---------------------------------------------------------------------------
// K3: rowwise softmax of S, emit bf16 attn.
// ---------------------------------------------------------------------------
__global__ __launch_bounds__(256)
void k_softmax(float* __restrict__ S, bf16* __restrict__ A) {
  __shared__ float red[256];
  const int r = blockIdx.x, tid = threadIdx.x;
  float* row = S + (size_t)r * L_TOT;
  float m = -1e30f;
  for (int j = tid; j < L_TOT; j += 256) m = fmaxf(m, row[j]);
  red[tid] = m; __syncthreads();
  for (int s = 128; s > 0; s >>= 1) { if (tid < s) red[tid] = fmaxf(red[tid], red[tid + s]); __syncthreads(); }
  m = red[0]; __syncthreads();
  float sum = 0.f;
  for (int j = tid; j < L_TOT; j += 256) { float e = __expf(row[j] - m); row[j] = e; sum += e; }
  red[tid] = sum; __syncthreads();
  for (int s = 128; s > 0; s >>= 1) { if (tid < s) red[tid] += red[tid + s]; __syncthreads(); }
  const float inv = 1.f / red[0];
  bf16* arow = A + (size_t)r * L_TOT;
  for (int j = tid; j < L_TOT; j += 256) arow[j] = (bf16)(row[j] * inv);
}

// ---------------------------------------------------------------------------
// K4: out[t,c] = sum_p attn[t,p] * g[c,p]  -> attno [L][512] bf16
// All tile staging is pure 16B-aligned copy -> fully async (ASYNCcnt engine).
// Per K-step: s_wait_asynccnt 0; barrier; issue next tile; 8 wmma.
// ---------------------------------------------------------------------------
__global__ __launch_bounds__(128)
void k_pv(const bf16* __restrict__ attn, const bf16* __restrict__ g,
          bf16* __restrict__ outp) {
  __shared__ bf16 As[2 * ASZ];
  __shared__ bf16 Bs[2 * BSZ];
  const int c0 = blockIdx.x * 128;
  const int t0 = blockIdx.y * 64;
  const int tid = threadIdx.x, lane = tid & 31, wave = tid >> 5;
  const int wmg = (wave >> 1) * 2, wng = (wave & 1) * 4;
  const int wm = wmg * 16, wn = wng * 16;
  v8f acc[2][4] = {};

  auto issueT = [&](int k, bf16* as, bf16* bs) {
#pragma unroll
    for (int i = 0; i < 2; ++i) {            // A: attn[t0+r][k+k8], row-major
      const int ch = i * 128 + tid, r = ch >> 2, k8 = (ch & 3) * 8;
      async_b128(as + a_sw(r, k8), attn + (size_t)(t0 + r) * L_TOT + k + k8);
    }
#pragma unroll
    for (int i = 0; i < 4; ++i) {            // B: g[c0+nc][k+k8] (k contiguous)
      const int ch = i * 128 + tid, nc = ch >> 2, k8 = (ch & 3) * 8;
      async_b128(bs + b_sw(k8, nc), g + (size_t)(c0 + nc) * L_TOT + k + k8);
    }
  };

  const int NK = L_TOT / 32;
  issueT(0, As, Bs);
  int cur = 0;
  for (int kt = 0; kt < NK; ++kt) {
    wait_async0();                 // my tile-kt copies landed in LDS
    __syncthreads();               // everyone's copies landed; buf cur^1 unread
    if (kt + 1 < NK) issueT((kt + 1) * 32, As + (cur ^ 1) * ASZ, Bs + (cur ^ 1) * BSZ);
    mma8(As + cur * ASZ, Bs + cur * BSZ, acc, wmg, wng, lane);
    cur ^= 1;
  }

  const int nn = lane & 15, mofs = (lane >> 4) * 8;
#pragma unroll
  for (int i = 0; i < 2; ++i)
#pragma unroll
    for (int j = 0; j < 4; ++j)
#pragma unroll
      for (int r = 0; r < 8; ++r) {
        const int t = t0 + wm + i * 16 + mofs + r;
        const int c = c0 + wn + j * 16 + nn;
        outp[(size_t)t * C_INNER + c] = (bf16)acc[i][j][r];
      }
}

// ---------------------------------------------------------------------------
// K5: p[n,d,l] = sum_c w_out[d,c] * out[n,l,c] + b_out[d] -> f32
// A (f32 weights) via VGPR staging + cvt; B via async copies.
// ---------------------------------------------------------------------------
__global__ __launch_bounds__(128)
void k_outproj(const bf16* __restrict__ outp, const float* __restrict__ W,
               const float* __restrict__ Bias, float* __restrict__ P) {
  __shared__ bf16 As[2 * ASZ];
  __shared__ bf16 Bs[2 * BSZ];
  const int l0 = blockIdx.x * 128;
  const int m0 = blockIdx.y * 64;
  const int n  = blockIdx.z;
  const bf16* ob = outp + (size_t)n * L_TOT * C_INNER;
  float* Pb = P + (size_t)n * C_DIM * L_TOT;
  const int tid = threadIdx.x, lane = tid & 31, wave = tid >> 5;
  const int wmg = (wave >> 1) * 2, wng = (wave & 1) * 4;
  const int wm = wmg * 16, wn = wng * 16;
  v8f acc[2][4] = {};

  const int rA = tid >> 3, cA = (tid & 7) * 4;
  float4 va[4];
  auto loadA = [&](int k) {
#pragma unroll
    for (int i = 0; i < 4; ++i)
      va[i] = *(const float4*)(W + (size_t)(m0 + i * 16 + rA) * C_INNER + k + cA);
  };
  auto storeA = [&](bf16* as) {
#pragma unroll
    for (int i = 0; i < 4; ++i) {
      bf16x4 t = { (bf16)va[i].x, (bf16)va[i].y, (bf16)va[i].z, (bf16)va[i].w };
      *(bf16x4*)(as + a_sw(i * 16 + rA, cA)) = t;
    }
  };
  auto issueB = [&](int k, bf16* bs) {
#pragma unroll
    for (int i = 0; i < 4; ++i) {            // B: out[l0+nl][k+k8] (k contiguous)
      const int ch = i * 128 + tid, nl = ch >> 2, k8 = (ch & 3) * 8;
      async_b128(bs + b_sw(k8, nl), ob + (size_t)(l0 + nl) * C_INNER + k + k8);
    }
  };

  const int NK = C_INNER / 32;
  loadA(0); issueB(0, Bs); storeA(As);
  int cur = 0;
  for (int kt = 0; kt < NK; ++kt) {
    wait_async0();
    __syncthreads();
    if (kt + 1 < NK) { issueB((kt + 1) * 32, Bs + (cur ^ 1) * BSZ); loadA((kt + 1) * 32); }
    mma8(As + cur * ASZ, Bs + cur * BSZ, acc, wmg, wng, lane);
    if (kt + 1 < NK) storeA(As + (cur ^ 1) * ASZ);
    cur ^= 1;
  }

  const int nn = lane & 15, mofs = (lane >> 4) * 8;
#pragma unroll
  for (int i = 0; i < 2; ++i)
#pragma unroll
    for (int j = 0; j < 4; ++j)
#pragma unroll
      for (int r = 0; r < 8; ++r) {
        const int d = m0 + wm + i * 16 + mofs + r;
        const int l = l0 + wn + j * 16 + nn;
        Pb[(size_t)d * L_TOT + l] = acc[i][j][r] + Bias[d];
      }
}

// ---------------------------------------------------------------------------
// K6/K7: deterministic two-stage mean/var over (C,T,H,W) per batch.
// ---------------------------------------------------------------------------
__global__ __launch_bounds__(256)
void k_rowstat(const float* __restrict__ P, float2* __restrict__ part) {
  __shared__ float s1[256], s2[256];
  const int d = blockIdx.x, n = blockIdx.y, tid = threadIdx.x;
  const float* row = P + ((size_t)n * C_DIM + d) * L_TOT;
  float a = 0.f, b = 0.f;
  for (int j = tid; j < L_TOT; j += 256) { const float v = row[j]; a += v; b += v * v; }
  s1[tid] = a; s2[tid] = b; __syncthreads();
  for (int s = 128; s > 0; s >>= 1) {
    if (tid < s) { s1[tid] += s1[tid + s]; s2[tid] += s2[tid + s]; }
    __syncthreads();
  }
  if (tid == 0) part[n * C_DIM + d] = make_float2(s1[0], s2[0]);
}

__global__ __launch_bounds__(256)
void k_stats(const float2* __restrict__ part, float* __restrict__ stats) {
  __shared__ float s1[256], s2[256];
  const int n = blockIdx.x, tid = threadIdx.x;
  float a = 0.f, b = 0.f;
  for (int j = tid; j < C_DIM; j += 256) { const float2 v = part[n * C_DIM + j]; a += v.x; b += v.y; }
  s1[tid] = a; s2[tid] = b; __syncthreads();
  for (int s = 128; s > 0; s >>= 1) {
    if (tid < s) { s1[tid] += s1[tid + s]; s2[tid] += s2[tid + s]; }
    __syncthreads();
  }
  if (tid == 0) {
    const float M = (float)C_DIM * (float)L_TOT;
    const float mu = s1[0] / M;
    const float var = s2[0] / M - mu * mu;
    stats[2 * n] = mu;
    stats[2 * n + 1] = rsqrtf(var + EPS_GN);
  }
}

// ---------------------------------------------------------------------------
// K8: y = x + (p - mu) * rstd
// ---------------------------------------------------------------------------
__global__ __launch_bounds__(256)
void k_final(const float* __restrict__ x, const float* __restrict__ P,
             const float* __restrict__ stats, float* __restrict__ y) {
  const size_t i = (size_t)blockIdx.x * 256 + threadIdx.x;
  const size_t tot = (size_t)N_B * C_DIM * L_TOT;
  if (i >= tot) return;
  const int n = (int)(i / ((size_t)C_DIM * L_TOT));
  y[i] = x[i] + (P[i] - stats[2 * n]) * stats[2 * n + 1];
}

// ---------------------------------------------------------------------------
extern "C" void kernel_launch(void* const* d_in, const int* in_sizes, int n_in,
                              void* d_out, int out_size, void* d_ws, size_t ws_size,
                              hipStream_t stream) {
  const float* x       = (const float*)d_in[0];
  const float* w_theta = (const float*)d_in[1];
  const float* b_theta = (const float*)d_in[2];
  const float* w_phi   = (const float*)d_in[3];
  const float* b_phi   = (const float*)d_in[4];
  const float* w_g     = (const float*)d_in[5];
  const float* b_g     = (const float*)d_in[6];
  const float* w_out   = (const float*)d_in[7];
  const float* b_out   = (const float*)d_in[8];

  char* ws = (char*)d_ws;
  size_t off = 0;
  auto alloc = [&](size_t bytes) -> void* {
    void* p = ws + off;
    off = (off + bytes + 255) & ~(size_t)255;
    return p;
  };
  bf16*   qkv   = (bf16*)  alloc((size_t)3 * N_B * C_INNER * L_TOT * sizeof(bf16));
  bf16*   attno = (bf16*)  alloc((size_t)N_B * L_TOT * C_INNER * sizeof(bf16));
  float*  P     = (float*) alloc((size_t)N_B * C_DIM * L_TOT * sizeof(float));
  float*  S     = (float*) alloc((size_t)L_TOT * L_TOT * sizeof(float));   // per-batch, reused
  bf16*   attn  = (bf16*)  alloc((size_t)L_TOT * L_TOT * sizeof(bf16));    // per-batch, reused
  float2* part  = (float2*)alloc((size_t)N_B * C_DIM * sizeof(float2));
  float*  stats = (float*) alloc(4 * sizeof(float));

  const dim3 blk(128);
  k_proj3<<<dim3(L_TOT / 128, C_INNER / 64, N_B * 3), blk, 0, stream>>>(
      x, w_theta, b_theta, w_phi, b_phi, w_g, b_g, qkv);

  for (int n = 0; n < N_B; ++n) {   // batches serialized so S/attn scratch is reused
    const bf16* th = qkv + ((size_t)0 * N_B + n) * C_INNER * L_TOT;
    const bf16* ph = qkv + ((size_t)1 * N_B + n) * C_INNER * L_TOT;
    const bf16* gg = qkv + ((size_t)2 * N_B + n) * C_INNER * L_TOT;
    k_scores<<<dim3(L_TOT / 128, L_TOT / 64), blk, 0, stream>>>(th, ph, S);
    k_softmax<<<dim3(L_TOT), dim3(256), 0, stream>>>(S, attn);
    k_pv<<<dim3(C_INNER / 128, L_TOT / 64), blk, 0, stream>>>(
        attn, gg, attno + (size_t)n * L_TOT * C_INNER);
  }

  k_outproj<<<dim3(L_TOT / 128, C_DIM / 64, N_B), blk, 0, stream>>>(attno, w_out, b_out, P);
  k_rowstat<<<dim3(C_DIM, N_B), dim3(256), 0, stream>>>(P, part);
  k_stats<<<dim3(N_B), dim3(256), 0, stream>>>(part, stats);
  const size_t tot = (size_t)N_B * C_DIM * L_TOT;
  k_final<<<dim3((unsigned)((tot + 255) / 256)), dim3(256), 0, stream>>>(
      x, P, stats, (float*)d_out);
}
